// KuramotoCoupling_87196426043474
// MI455X (gfx1250) — matrix operand order repaired and checked by reading, not compiled, not run on hardware
//
#include <hip/hip_runtime.h>

typedef unsigned short u16;
typedef __attribute__((ext_vector_type(16))) __bf16 v16bf;
typedef __attribute__((ext_vector_type(8)))  float   v8f;

union Frag32 { v16bf v; uint4 q[2]; };

__device__ __forceinline__ u16 f2bf(float x) {
  unsigned u = __float_as_uint(x);
  u = (u + 0x7FFFu + ((u >> 16) & 1u)) >> 16;
  return (u16)u;
}

#define B_  2
#define S_  2048
#define D_  1024
#define H_  16
#define HD_ 64

// ---------------------------------------------------------------------------
// Tensor Data Mover (TDM) support, probe-guarded. 2-D tile load with LDS
// padding: rows of 64B data + 16B pad -> matches u16[128][40] LDS layout.
// ---------------------------------------------------------------------------
#if defined(__gfx1250__) && __has_builtin(__builtin_amdgcn_tensor_load_to_lds)
#define HAVE_TDM 1
typedef __attribute__((ext_vector_type(4))) unsigned int u32x4;
typedef __attribute__((ext_vector_type(8))) int          i32x8;
typedef __attribute__((ext_vector_type(4))) int          i32x4;

__device__ __forceinline__ void tdm_load_tile_2d(
    const u16* gsrc,          // global address of tile start
    unsigned   ldsAddr,       // LDS byte address of tile destination
    unsigned   tensorD0,      // tensor dim0 length (elements)
    unsigned   tensorD1,      // tensor dim1 length (elements)
    unsigned   tileD0,        // tile width  (elements) == 32
    unsigned   tileD1,        // tile height (rows)     == 128
    unsigned   stride0)       // row stride (elements)
{
  unsigned long long ga = (unsigned long long)(size_t)gsrc;
  // D# group 0: count=1 | lds_addr | global_addr[56:0] | type=2
  u32x4 g0 = { 1u,
               ldsAddr,
               (unsigned)(ga & 0xFFFFFFFFu),
               ((unsigned)((ga >> 32) & 0x01FFFFFFu)) | 0x80000000u };
  // D# group 1: data_size=2B (code 1), pad_enable, pad_interval=16 DWORDs
  // (code 3), pad_amount=4 DWORDs (code 3); dims/strides per ISA 8.4 packing.
  i32x8 g1 = {
    (int)((1u << 16) | (1u << 20) | (3u << 22) | (3u << 25)),
    (int)((tensorD0 & 0xFFFFu) << 16),
    (int)(((tensorD0 >> 16) & 0xFFFFu) | ((tensorD1 & 0xFFFFu) << 16)),
    (int)(((tensorD1 >> 16) & 0xFFFFu) | ((tileD0 & 0xFFFFu) << 16)),
    (int)(tileD1 & 0xFFFFu),
    (int)stride0,
    0, 0 };
  i32x4 gz = { 0, 0, 0, 0 };
#if __clang_major__ >= 23
  i32x8 gz8 = { 0, 0, 0, 0, 0, 0, 0, 0 };
  __builtin_amdgcn_tensor_load_to_lds(g0, g1, gz, gz, gz8, 0);
#else
  __builtin_amdgcn_tensor_load_to_lds(g0, g1, gz, gz, 0);
#endif
}
#else
#define HAVE_TDM 0
#endif

// ---------------------------------------------------------------------------
// Prep: mag = |z| (bf16), sin(phase)=zi/|z|, cos(phase)=zr/|z| (bf16)
// ---------------------------------------------------------------------------
__global__ __launch_bounds__(256) void prep_kernel(
    const float* __restrict__ zre, const float* __restrict__ zim,
    u16* __restrict__ magB, u16* __restrict__ sinB, u16* __restrict__ cosB,
    int n)
{
  int i = blockIdx.x * 256 + threadIdx.x;
  if (i >= n) return;
  float zr = zre[i], zi = zim[i];
  float m = sqrtf(zr * zr + zi * zi);
  float sp = 0.f, cp = 1.f;
  if (m > 0.f) { float inv = 1.f / m; sp = zi * inv; cp = zr * inv; }
  magB[i] = f2bf(m);
  sinB[i] = f2bf(sp);
  cosB[i] = f2bf(cp);
}

// ---------------------------------------------------------------------------
// f32 -> bf16 convert (weights)
// ---------------------------------------------------------------------------
__global__ __launch_bounds__(256) void cvt_kernel(
    const float* __restrict__ src, u16* __restrict__ dst, int n)
{
  int i = blockIdx.x * 256 + threadIdx.x;
  if (i < n) dst[i] = f2bf(src[i]);
}

// ---------------------------------------------------------------------------
// GEMM: Y[m,n] = sum_k A[m,k] * W[n,k]   (i.e. A @ W^T), all bf16 in, f32 acc
// Block tile 128x128, 256 threads = 8 waves in 2(M) x 4(N), wave tile 64x32.
// Tile staging via TDM DMA (one descriptor per tile, LDS padding in-DMA)
// when available, else manual b128 global->LDS copies.
// ---------------------------------------------------------------------------
__global__ __launch_bounds__(256) void gemm_bf16_kernel(
    const u16* __restrict__ A, const u16* __restrict__ W,
    u16* __restrict__ outB, float* __restrict__ outF,
    int N, int K)
{
  __shared__ __align__(16) u16 As[128][40];
  __shared__ __align__(16) u16 Ws[128][40];
  const int t = threadIdx.x;
  const int lane = t & 31, w = t >> 5;
  const int wm = w & 1, wn = w >> 1;
  const int g = lane >> 4, ln = lane & 15;
  const int m0 = blockIdx.x * 128, n0 = blockIdx.y * 128;

  v8f zero = {};
  v8f acc[4][2];
#pragma unroll
  for (int i = 0; i < 4; ++i)
#pragma unroll
    for (int j = 0; j < 2; ++j) acc[i][j] = zero;

  for (int k0 = 0; k0 < K; k0 += 32) {
    __syncthreads();

    // prefetch next K tile into L2 (global_prefetch_b8)
    if (k0 + 32 < K) {
      int pr = t >> 1;
      if (t & 1) __builtin_prefetch(&W[(size_t)(n0 + pr) * K + k0 + 32], 0, 1);
      else       __builtin_prefetch(&A[(size_t)(m0 + pr) * K + k0 + 32], 0, 1);
    }

#if HAVE_TDM
    // one DMA per tile: wave0 -> A tile, wave1 -> W tile (EXEC ignored, one
    // TDM issue per wave); issuing waves drain TENSORcnt before the barrier.
    if (w == 0)
      tdm_load_tile_2d(&A[(size_t)m0 * K + k0],
                       (unsigned)(size_t)(void*)&As[0][0],
                       (unsigned)K, 1u << 24, 32u, 128u, (unsigned)K);
    if (w == 1)
      tdm_load_tile_2d(&W[(size_t)n0 * K + k0],
                       (unsigned)(size_t)(void*)&Ws[0][0],
                       (unsigned)K, 1u << 24, 32u, 128u, (unsigned)K);
    __builtin_amdgcn_s_wait_tensorcnt(0);
#else
    // stage 128x32 tiles of A and W (512 x 16B chunks each, 2 per thread)
#pragma unroll
    for (int i = 0; i < 2; ++i) {
      int idx = t + i * 256;
      int row = idx >> 2, c = (idx & 3) * 8;
      *(uint4*)&As[row][c] = *(const uint4*)&A[(size_t)(m0 + row) * K + k0 + c];
      *(uint4*)&Ws[row][c] = *(const uint4*)&W[(size_t)(n0 + row) * K + k0 + c];
    }
#endif
    __syncthreads();

    Frag32 af[4], bf[2];
#pragma unroll
    for (int ms = 0; ms < 4; ++ms) {
      const u16* p = &As[wm * 64 + ms * 16 + ln][g * 16];
      af[ms].q[0] = *(const uint4*)(p);
      af[ms].q[1] = *(const uint4*)(p + 8);
    }
#pragma unroll
    for (int ns = 0; ns < 2; ++ns) {
      const u16* p = &Ws[wn * 32 + ns * 16 + ln][g * 16];
      bf[ns].q[0] = *(const uint4*)(p);
      bf[ns].q[1] = *(const uint4*)(p + 8);
    }
#pragma unroll
    for (int ms = 0; ms < 4; ++ms)
#pragma unroll
      for (int ns = 0; ns < 2; ++ns)
        acc[ms][ns] = __builtin_amdgcn_wmma_f32_16x16x32_bf16(
            false, af[ms].v, false, bf[ns].v, (short)0, acc[ms][ns],
            false, false);
  }

#pragma unroll
  for (int ms = 0; ms < 4; ++ms)
#pragma unroll
    for (int ns = 0; ns < 2; ++ns)
#pragma unroll
      for (int r = 0; r < 8; ++r) {
        int row = m0 + wm * 64 + ms * 16 + r + 8 * g;
        int col = n0 + wn * 32 + ns * 16 + ln;
        float v = acc[ms][ns][r];
        if (outB) outB[(size_t)row * N + col] = f2bf(v);
        else      outF[(size_t)row * N + col] = v;
      }
}

// ---------------------------------------------------------------------------
// Flash attention with 3 value streams (v, sin, cos) + fused Kuramoto epilogue
// grid: (S/64, B*H); 128 threads = 4 waves, each wave owns 16 query rows.
// ---------------------------------------------------------------------------
__global__ __launch_bounds__(128) void attn_kernel(
    const u16* __restrict__ qb, const u16* __restrict__ kb,
    const u16* __restrict__ vb, const u16* __restrict__ snb,
    const u16* __restrict__ csb,
    const float* __restrict__ zre, const float* __restrict__ zim,
    const float* __restrict__ coupling,
    u16* __restrict__ voutB, float* __restrict__ outPhase)
{
  __shared__ __align__(16) u16 Ks[64][72];          // [kk][d]  row-major
  __shared__ __align__(16) u16 Vt[64][72];          // [d][kk]  transposed
  __shared__ __align__(16) u16 St[64][72];
  __shared__ __align__(16) u16 Ct[64][72];
  __shared__ __align__(16) u16 Ps[4][16][72];       // per-wave P tile 16x64

  const int t = threadIdx.x, lane = t & 31, w = t >> 5;
  const int g = lane >> 4, ln = lane & 15;
  const int q0 = blockIdx.x * 64;
  const int bh = blockIdx.y, b = bh >> 4, h = bh & 15;

  // Q fragments (A layout) for this wave's 16 rows, K = HD = 64 -> 2 steps
  Frag32 qf[2];
  {
    const u16* qrow = qb + (size_t)(b * S_ + q0 + w * 16 + ln) * D_ + h * HD_;
#pragma unroll
    for (int kk2 = 0; kk2 < 2; ++kk2) {
      qf[kk2].q[0] = *(const uint4*)&qrow[kk2 * 32 + g * 16];
      qf[kk2].q[1] = *(const uint4*)&qrow[kk2 * 32 + g * 16 + 8];
    }
  }

  v8f zero = {};
  v8f oV[4], oS[4], oC[4];
#pragma unroll
  for (int ns = 0; ns < 4; ++ns) { oV[ns] = zero; oS[ns] = zero; oC[ns] = zero; }
  float m[8], ls[8];
#pragma unroll
  for (int r = 0; r < 8; ++r) { m[r] = -1.0e30f; ls[r] = 0.f; }

  for (int kc = 0; kc < S_; kc += 64) {
    __syncthreads();
    // K tile: 64 rows x 64 halves = 512 16B chunks, row-major
#pragma unroll
    for (int i = 0; i < 4; ++i) {
      int idx = t + i * 128;
      int r = idx >> 3, c = (idx & 7) * 8;
      *(uint4*)&Ks[r][c] =
          *(const uint4*)&kb[(size_t)(b * S_ + kc + r) * D_ + h * HD_ + c];
    }
    // v / sin / cos tiles: transposed into LDS ([d][kk])
#pragma unroll
    for (int i = 0; i < 4; ++i) {
      int idx = t + i * 128;
      int r = idx >> 3, c = (idx & 7) * 8;
      size_t go = (size_t)(b * S_ + kc + r) * D_ + h * HD_ + c;
      uint4 xv = *(const uint4*)&vb[go];
      uint4 xs = *(const uint4*)&snb[go];
      uint4 xc = *(const uint4*)&csb[go];
      const u16* pv = (const u16*)&xv;
      const u16* ps = (const u16*)&xs;
      const u16* pc = (const u16*)&xc;
#pragma unroll
      for (int j = 0; j < 8; ++j) {
        Vt[c + j][r] = pv[j];
        St[c + j][r] = ps[j];
        Ct[c + j][r] = pc[j];
      }
    }
    __syncthreads();

    // scores: S = Q(16x64) . K^T  -> 4 column subtiles of 16
    v8f sAcc[4];
#pragma unroll
    for (int ns = 0; ns < 4; ++ns) sAcc[ns] = zero;
#pragma unroll
    for (int kk2 = 0; kk2 < 2; ++kk2)
#pragma unroll
      for (int ns = 0; ns < 4; ++ns) {
        Frag32 bf;
        const u16* p = &Ks[ns * 16 + ln][kk2 * 32 + g * 16];
        bf.q[0] = *(const uint4*)(p);
        bf.q[1] = *(const uint4*)(p + 8);
        sAcc[ns] = __builtin_amdgcn_wmma_f32_16x16x32_bf16(
            false, qf[kk2].v, false, bf.v, (short)0, sAcc[ns], false, false);
      }

    // online softmax per row (row = r + 8*g; N across 16 lanes of each group)
#pragma unroll
    for (int r = 0; r < 8; ++r) {
      float s0 = sAcc[0][r] * 0.125f, s1 = sAcc[1][r] * 0.125f;
      float s2 = sAcc[2][r] * 0.125f, s3 = sAcc[3][r] * 0.125f;
      sAcc[0][r] = s0; sAcc[1][r] = s1; sAcc[2][r] = s2; sAcc[3][r] = s3;
      float mx = fmaxf(fmaxf(s0, s1), fmaxf(s2, s3));
#pragma unroll
      for (int msk = 1; msk < 16; msk <<= 1)
        mx = fmaxf(mx, __shfl_xor(mx, msk, 32));
      float mnew = fmaxf(m[r], mx);
      float alpha = __expf(m[r] - mnew);
      m[r] = mnew;
      ls[r] *= alpha;
#pragma unroll
      for (int ns = 0; ns < 4; ++ns) {
        oV[ns][r] *= alpha; oS[ns][r] *= alpha; oC[ns][r] *= alpha;
      }
      float pr = 0.f;
#pragma unroll
      for (int ns = 0; ns < 4; ++ns) {
        float pe = __expf(sAcc[ns][r] - mnew);
        pr += pe;
        Ps[w][r + 8 * g][ns * 16 + ln] = f2bf(pe);
      }
#pragma unroll
      for (int msk = 1; msk < 16; msk <<= 1)
        pr += __shfl_xor(pr, msk, 32);
      ls[r] += pr;
    }
    __syncthreads();

    // O += P(16x64) x {V,sin,cos}(64x64); B-frags from transposed tiles
#pragma unroll
    for (int kk2 = 0; kk2 < 2; ++kk2) {
      Frag32 pa;
      const u16* pp = &Ps[w][ln][kk2 * 32 + g * 16];
      pa.q[0] = *(const uint4*)(pp);
      pa.q[1] = *(const uint4*)(pp + 8);
#pragma unroll
      for (int ns = 0; ns < 4; ++ns) {
        Frag32 bv, bs, bc;
        const u16* pv = &Vt[ns * 16 + ln][kk2 * 32 + g * 16];
        const u16* ps = &St[ns * 16 + ln][kk2 * 32 + g * 16];
        const u16* pc = &Ct[ns * 16 + ln][kk2 * 32 + g * 16];
        bv.q[0] = *(const uint4*)(pv); bv.q[1] = *(const uint4*)(pv + 8);
        bs.q[0] = *(const uint4*)(ps); bs.q[1] = *(const uint4*)(ps + 8);
        bc.q[0] = *(const uint4*)(pc); bc.q[1] = *(const uint4*)(pc + 8);
        oV[ns] = __builtin_amdgcn_wmma_f32_16x16x32_bf16(
            false, pa.v, false, bv.v, (short)0, oV[ns], false, false);
        oS[ns] = __builtin_amdgcn_wmma_f32_16x16x32_bf16(
            false, pa.v, false, bs.v, (short)0, oS[ns], false, false);
        oC[ns] = __builtin_amdgcn_wmma_f32_16x16x32_bf16(
            false, pa.v, false, bc.v, (short)0, oC[ns], false, false);
      }
    }
  }

  // epilogue: normalize, write value_out (bf16 for Wo GEMM),
  // fused Kuramoto phase update (recompute sin/cos in f32 from z)
  float cstr = coupling[0];
#pragma unroll
  for (int r = 0; r < 8; ++r) {
    float inv = 1.0f / ls[r];
    int qrow = q0 + w * 16 + r + 8 * g;
    size_t base = (size_t)(b * S_ + qrow) * D_ + h * HD_;
#pragma unroll
    for (int ns = 0; ns < 4; ++ns) {
      int col = ns * 16 + ln;
      float ov = oV[ns][r] * inv;
      float os = oS[ns][r] * inv;
      float oc = oC[ns][r] * inv;
      voutB[base + col] = f2bf(ov);
      float zr = zre[base + col], zi = zim[base + col];
      float mg = sqrtf(zr * zr + zi * zi);
      float sp = 0.f, cp = 1.f;
      if (mg > 0.f) { float im = 1.f / mg; sp = zi * im; cp = zr * im; }
      outPhase[base + col] = cstr * (cp * os - sp * oc);
    }
  }
}

// ---------------------------------------------------------------------------
extern "C" void kernel_launch(void* const* d_in, const int* in_sizes, int n_in,
                              void* d_out, int out_size, void* d_ws,
                              size_t ws_size, hipStream_t stream)
{
  (void)in_sizes; (void)n_in; (void)out_size; (void)ws_size;
  const float* zre = (const float*)d_in[0];
  const float* zim = (const float*)d_in[1];
  const float* Wq  = (const float*)d_in[2];
  const float* Wk  = (const float*)d_in[3];
  const float* Wv  = (const float*)d_in[4];
  const float* Wo  = (const float*)d_in[5];
  const float* cs  = (const float*)d_in[6];
  float* out = (float*)d_out;

  const int    BS = B_ * S_;              // 4096 rows
  const size_t NM = (size_t)BS * D_;      // 4,194,304 elements

  char* ws = (char*)d_ws;
  const size_t MB = 1024 * 1024;
  u16* magB = (u16*)(ws);
  u16* qB   = (u16*)(ws +  8 * MB);
  u16* kB   = (u16*)(ws + 16 * MB);
  u16* vB   = (u16*)(ws + 24 * MB);
  u16* snB  = (u16*)(ws + 32 * MB);
  u16* csB  = (u16*)(ws + 40 * MB);
  u16* voB  = (u16*)(ws + 48 * MB);
  u16* WqB  = (u16*)(ws + 56 * MB);
  u16* WkB  = (u16*)(ws + 58 * MB);
  u16* WvB  = (u16*)(ws + 60 * MB);
  u16* WoB  = (u16*)(ws + 62 * MB);

  prep_kernel<<<(int)(NM / 256), 256, 0, stream>>>(zre, zim, magB, snB, csB,
                                                   (int)NM);
  cvt_kernel<<<(D_ * D_) / 256, 256, 0, stream>>>(Wq, WqB, D_ * D_);
  cvt_kernel<<<(D_ * D_) / 256, 256, 0, stream>>>(Wk, WkB, D_ * D_);
  cvt_kernel<<<(D_ * D_) / 256, 256, 0, stream>>>(Wv, WvB, D_ * D_);
  cvt_kernel<<<(D_ * D_) / 256, 256, 0, stream>>>(Wo, WoB, D_ * D_);

  dim3 gg(BS / 128, D_ / 128);     // (32, 8)
  gemm_bf16_kernel<<<gg, 256, 0, stream>>>(magB, WqB, qB, nullptr, D_, D_);
  gemm_bf16_kernel<<<gg, 256, 0, stream>>>(magB, WkB, kB, nullptr, D_, D_);
  gemm_bf16_kernel<<<gg, 256, 0, stream>>>(magB, WvB, vB, nullptr, D_, D_);

  dim3 ga(S_ / 64, B_ * H_);       // (32, 32)
  attn_kernel<<<ga, 128, 0, stream>>>(qB, kB, vB, snB, csB, zre, zim, cs,
                                      voB, out);

  gemm_bf16_kernel<<<gg, 256, 0, stream>>>(voB, WoB, nullptr, out + NM,
                                           D_, D_);
}
